// MixtureOfDepths_18021682774286
// MI455X (gfx1250) — matrix operand order repaired and checked by reading, not compiled, not run on hardware
//
#include <hip/hip_runtime.h>
#include <hip/hip_bf16.h>

typedef __attribute__((ext_vector_type(16))) _Float16 v16h;
typedef __attribute__((ext_vector_type(8)))  float    v8f;

#define B_   2
#define S_   1024
#define H_   768
#define NH_  12
#define L_   4
#define I_   3072
#define V_   50257
#define HD_  64
#define CAP_ 512
#define M_   (B_*S_)      /* 2048 token rows */
#define EPS_ 1e-6f

#define WMMA_F16(Af,Bf,Cf) \
  __builtin_amdgcn_wmma_f32_16x16x32_f16(false,(Af),false,(Bf),(short)0,(Cf),false,false)

// ---------------------------------------------------------------------------
// Embedding gather + positional add: x[b,s,h] = embed[ids[b,s],h] + pos[s,h]
// ---------------------------------------------------------------------------
__global__ void embed_k(const float* __restrict__ emb, const float* __restrict__ pos,
                        const int* __restrict__ ids, float* __restrict__ X) {
  size_t i = (size_t)blockIdx.x * blockDim.x + threadIdx.x;
  if (i >= (size_t)M_ * H_) return;
  int hcol = (int)(i % H_);
  int srow = (int)((i / H_) % S_);
  int b    = (int)(i / ((size_t)S_ * H_));
  int tok  = ids[b * S_ + srow];
  X[i] = emb[(size_t)tok * H_ + hcol] + pos[(size_t)srow * H_ + hcol];
}

// ---------------------------------------------------------------------------
// Router logits: rl[row] = dot(x[row,:], w[:])   (one 256-thread block / row)
// ---------------------------------------------------------------------------
__global__ void __launch_bounds__(256)
rowdot_k(const float* __restrict__ X, const float* __restrict__ w,
         float* __restrict__ rl) {
  __shared__ float red[8];
  int row = blockIdx.x;
  int t = threadIdx.x;
  const float* xr = X + (size_t)row * H_;
  float s = xr[t] * w[t] + xr[t + 256] * w[t + 256] + xr[t + 512] * w[t + 512];
  #pragma unroll
  for (int m = 16; m >= 1; m >>= 1) s += __shfl_xor(s, m, 32);
  if ((t & 31) == 0) red[t >> 5] = s;
  __syncthreads();
  if (t == 0) {
    float tot = 0.f;
    #pragma unroll
    for (int i = 0; i < 8; ++i) tot += red[i];
    rl[row] = tot;
  }
}

// ---------------------------------------------------------------------------
// Top-CAP mask via rank counting; matches jax.lax.top_k tie-breaking.
// Forward value of (mask + probs - stop_grad(probs)) is exactly mask.
// ---------------------------------------------------------------------------
__global__ void topk_mask_k(const float* __restrict__ rl, float* __restrict__ mask) {
  int idx = blockIdx.x * blockDim.x + threadIdx.x;
  if (idx >= M_) return;
  int b = idx / S_, s = idx % S_;
  const float* row = rl + b * S_;
  float my = row[s];
  int rank = 0;
  for (int t = 0; t < S_; ++t) {
    float v = row[t];
    rank += (v > my) || (v == my && t < s);
  }
  mask[idx] = (rank < CAP_) ? 1.f : 0.f;
}

// ---------------------------------------------------------------------------
// LayerNorm over H=768, fp32 stats, f16 output (consumed only as WMMA A-frags)
// ---------------------------------------------------------------------------
__global__ void __launch_bounds__(256)
layernorm_k(const float* __restrict__ X, const float* __restrict__ gam,
            const float* __restrict__ bet, _Float16* __restrict__ Y) {
  __shared__ float red[8];
  int row = blockIdx.x;
  int t = threadIdx.x;
  const float* xr = X + (size_t)row * H_;
  float v0 = xr[t], v1 = xr[t + 256], v2 = xr[t + 512];
  float s = v0 + v1 + v2;
  #pragma unroll
  for (int m = 16; m >= 1; m >>= 1) s += __shfl_xor(s, m, 32);
  if ((t & 31) == 0) red[t >> 5] = s;
  __syncthreads();
  float tot = 0.f;
  #pragma unroll
  for (int i = 0; i < 8; ++i) tot += red[i];
  float mean = tot * (1.f / (float)H_);
  __syncthreads();
  float d0 = v0 - mean, d1 = v1 - mean, d2 = v2 - mean;
  float sq = d0 * d0 + d1 * d1 + d2 * d2;
  #pragma unroll
  for (int m = 16; m >= 1; m >>= 1) sq += __shfl_xor(sq, m, 32);
  if ((t & 31) == 0) red[t >> 5] = sq;
  __syncthreads();
  float tot2 = 0.f;
  #pragma unroll
  for (int i = 0; i < 8; ++i) tot2 += red[i];
  float inv = rsqrtf(tot2 * (1.f / (float)H_) + EPS_);
  _Float16* yr = Y + (size_t)row * H_;
  yr[t]       = (_Float16)(d0 * inv * gam[t]       + bet[t]);
  yr[t + 256] = (_Float16)(d1 * inv * gam[t + 256] + bet[t + 256]);
  yr[t + 512] = (_Float16)(d2 * inv * gam[t + 512] + bet[t + 512]);
}

// ---------------------------------------------------------------------------
// SwiGLU elementwise in f16 storage: f = silu(g) * u   (g may alias f)
// ---------------------------------------------------------------------------
__global__ void swiglu_k(const _Float16* __restrict__ g, const _Float16* __restrict__ u,
                         _Float16* __restrict__ f, int n) {
  int i = blockIdx.x * blockDim.x + threadIdx.x;
  if (i >= n) return;
  float gv = (float)g[i];
  f[i] = (_Float16)((gv / (1.f + __expf(-gv))) * (float)u[i]);
}

// ---------------------------------------------------------------------------
// WMMA GEMM, compile-time shapes: C[2048,N] = A[2048,K] @ Bm[K,N].
// A is f16 in memory (produced once by LN/attention/SwiGLU) -> A-frag loads
// are straight 16B vector loads, no in-loop cvt, half the A-side L2 traffic.
// B (weights) is fp32, converted in-loop.
// One wave computes a 64x64 C block (MT=4 x NT=4): each A fragment feeds 4
// WMMAs and each B fragment feeds 4 WMMAs, balancing A/B L2 traffic and
// giving 16 independent accumulators that hide WMMA->WMMA RAW hazards.
// Block mapping: the 4 waves of a block take consecutive M-groups of the SAME
// N-group, so all 4 issue an identical (expensive, strided fp32) B address
// stream -- the WGP cache serves 3 of 4 waves, cutting per-block traffic from
// A + 4B to 4A + B (A panels are cheap vectorized f16).
// OUT16: plain f16 store (intermediates). Else fp32 store, optionally FUSE:
// C[m,n] += rowScale[m] * acc (routed residual add).
// For the ragged lm_head N, edge lanes clamp their load column (a clamped
// lane's B column only feeds its own output column, masked at store time).
// ---------------------------------------------------------------------------
template <int N, int K, bool FUSE, bool OUT16>
__global__ void __launch_bounds__(128)
gemm_wmma(const _Float16* __restrict__ A, const float* __restrict__ Bm,
          float* __restrict__ Cf, _Float16* __restrict__ Ch,
          const float* __restrict__ rowScale) {
  constexpr int MT = 4;                       // M-tiles per wave (64 rows)
  constexpr int NT = 4;                       // N-tiles per wave (64 cols)
  constexpr int TnT = (N + 15) / 16;          // 16-col tiles
  constexpr int NG  = (TnT + NT - 1) / NT;    // 64-col groups
  constexpr int MG  = M_ / (16 * MT);         // 64-row groups
  constexpr bool EDGE = (N % (16 * NT)) != 0; // only the lm_head instantiation
  int wave = threadIdx.x >> 5;
  int lane = threadIdx.x & 31;
  int half = lane >> 4;
  int l16  = lane & 15;
  int unit = blockIdx.x * 4 + wave;
  if (unit >= MG * NG) return;                // wave-uniform exit (EXEC full)
  int mg = unit % MG, ng = unit / MG;         // waves in a block share ng => share B

  const _Float16* aPtr[MT];
  #pragma unroll
  for (int t = 0; t < MT; ++t)
    aPtr[t] = A + (size_t)(mg * 64 + t * 16 + l16) * K + half * 8;

  int nColS[NT];                              // store columns
  const float* bPtr[NT];
  #pragma unroll
  for (int j = 0; j < NT; ++j) {
    int nc = ng * 64 + j * 16 + l16;
    nColS[j] = nc;
    int ncLd = nc;
    if constexpr (EDGE) ncLd = (nc < N) ? nc : (N - 1);
    bPtr[j] = Bm + (size_t)(half * 8) * N + ncLd;
  }

  v8f acc[MT][NT];
  #pragma unroll
  for (int t = 0; t < MT; ++t)
    #pragma unroll
    for (int j = 0; j < NT; ++j)
      #pragma unroll
      for (int r = 0; r < 8; ++r) acc[t][j][r] = 0.f;

  for (int k0 = 0; k0 < K; k0 += 32) {
    v16h bf[NT];                              // B frag: lane col = l16; k(e) = (e>>3)*16 + half*8 + (e&7)
    #pragma unroll
    for (int j = 0; j < NT; ++j) {
      #pragma unroll
      for (int e = 0; e < 16; ++e) bf[j][e] = (_Float16)bPtr[j][(size_t)((e >> 3) * 16 + (e & 7)) * N];
      bPtr[j] += (size_t)32 * N;
    }
    #pragma unroll
    for (int t = 0; t < MT; ++t) {            // A frag: lane row = l16; k(e)=(e>>3)*16+half*8+(e&7)
      v16h af;
      #pragma unroll
      for (int g = 0; g < 2; ++g)
        #pragma unroll
        for (int i = 0; i < 8; ++i) af[g * 8 + i] = aPtr[t][g * 16 + i];
      aPtr[t] += 32;
      #pragma unroll
      for (int j = 0; j < NT; ++j)
        acc[t][j] = WMMA_F16(af, bf[j], acc[t][j]);
    }
  }

  #pragma unroll
  for (int j = 0; j < NT; ++j) {
    bool nOK = true;
    if constexpr (EDGE) nOK = (nColS[j] < N);
    if (nOK) {
      #pragma unroll
      for (int t = 0; t < MT; ++t)
        #pragma unroll
        for (int r = 0; r < 8; ++r) {
          int m = mg * 64 + t * 16 + half * 8 + r;   // C/D layout: m = vgpr + half*8
          size_t idx = (size_t)m * N + nColS[j];
          if constexpr (OUT16)     Ch[idx] = (_Float16)acc[t][j][r];
          else if constexpr (FUSE) Cf[idx] += rowScale[m] * acc[t][j][r];
          else                     Cf[idx] = acc[t][j][r];
        }
    }
  }
}

#define GEMM_LAUNCH(Nv, Kv, Fv, O16v, Ap, Bp, Cfp, Chp, Rp)                      \
  do {                                                                           \
    constexpr int TnT_ = ((Nv) + 15) / 16;                                       \
    constexpr int NG_  = (TnT_ + 3) / 4;                                         \
    constexpr int units_ = (M_ / 64) * NG_;                                      \
    gemm_wmma<(Nv), (Kv), (Fv), (O16v)><<<(units_ + 3) / 4, 128, 0, stream>>>(   \
        (Ap), (Bp), (Cfp), (Chp), (Rp));                                         \
  } while (0)

// ---------------------------------------------------------------------------
// Flash attention, f16 in / f16 out: one wave per (b, head, 16-query tile).
// Online softmax over 32-key chunks. S = Q@K^T via WMMA; P staged through LDS
// (C/D -> A layout); O += P@V via WMMA. Causal mask applied pre-softmax.
// Q,K,V,O: [B*S, H] f16 row-major, head h at columns [h*64, h*64+64).
// ---------------------------------------------------------------------------
__global__ void __launch_bounds__(128)
flash_attn_k(const _Float16* __restrict__ Q, const _Float16* __restrict__ Kc,
             const _Float16* __restrict__ Vc, _Float16* __restrict__ O) {
  __shared__ _Float16 psh[4][16 * 32];
  int wave = threadIdx.x >> 5;
  int lane = threadIdx.x & 31;
  int half = lane >> 4;
  int l16  = lane & 15;
  int unit = blockIdx.x * 4 + wave;      // (b*NH + h)*(S/16) + qt
  int qt = unit & 63;                    // S/16 == 64
  int bh = unit >> 6;
  int h  = bh % NH_;
  int b  = bh / NH_;
  const float scale = 0.125f;            // HD^-0.5
  size_t rowBase = (size_t)b * S_;

  // Q A-fragments for the two 32-wide K-dim steps (dims 0..31 and 32..63)
  v16h qa0, qa1;
  {
    const _Float16* qp = Q + (rowBase + qt * 16 + l16) * H_ + h * HD_;
    #pragma unroll
    for (int g = 0; g < 2; ++g)
      #pragma unroll
      for (int i = 0; i < 8; ++i) {
        qa0[g * 8 + i] = qp[g * 16 + half * 8 + i];
        qa1[g * 8 + i] = qp[32 + g * 16 + half * 8 + i];
      }
  }
  v8f oacc[4];
  #pragma unroll
  for (int t = 0; t < 4; ++t)
    #pragma unroll
    for (int r = 0; r < 8; ++r) oacc[t][r] = 0.f;
  float mrun[8], lrun[8];
  #pragma unroll
  for (int r = 0; r < 8; ++r) { mrun[r] = -1e30f; lrun[r] = 0.f; }

  int qTop = qt * 16 + 15;
  int nChunk = (qTop >> 5) + 1;          // 32-key chunks touching the causal region
  for (int c = 0; c < nChunk; ++c) {
    int kBase = c * 32;
    v8f s0 = {}, s1 = {};
    // scores for keys [kBase, kBase+16) and [kBase+16, kBase+32)
    #pragma unroll
    for (int sub = 0; sub < 2; ++sub) {
      const _Float16* kp = Kc + (rowBase + kBase + sub * 16 + l16) * H_ + h * HD_;
      v16h bf0, bf1;
      #pragma unroll
      for (int e = 0; e < 16; ++e) {
        bf0[e] = kp[(e >> 3) * 16 + half * 8 + (e & 7)];        // dims 0..31
        bf1[e] = kp[32 + (e >> 3) * 16 + half * 8 + (e & 7)];   // dims 32..63
      }
      if (sub == 0) { s0 = WMMA_F16(qa0, bf0, s0); s0 = WMMA_F16(qa1, bf1, s0); }
      else          { s1 = WMMA_F16(qa0, bf0, s1); s1 = WMMA_F16(qa1, bf1, s1); }
    }
    // scale + causal mask + online softmax bookkeeping (rows live per-16-lane-group)
    float fac[8];
    #pragma unroll
    for (int r = 0; r < 8; ++r) {
      int qrow = qt * 16 + half * 8 + r;
      int key0 = kBase + l16;
      int key1 = kBase + 16 + l16;
      s0[r] = (key0 <= qrow) ? s0[r] * scale : -1e30f;
      s1[r] = (key1 <= qrow) ? s1[r] * scale : -1e30f;
      float cm = fmaxf(s0[r], s1[r]);
      #pragma unroll
      for (int m = 8; m >= 1; m >>= 1) cm = fmaxf(cm, __shfl_xor(cm, m, 32));
      float nm = fmaxf(mrun[r], cm);
      fac[r] = __expf(mrun[r] - nm);
      mrun[r] = nm;
      float p0 = __expf(s0[r] - nm);
      float p1 = __expf(s1[r] - nm);
      s0[r] = p0; s1[r] = p1;
      float rs = p0 + p1;
      #pragma unroll
      for (int m = 8; m >= 1; m >>= 1) rs += __shfl_xor(rs, m, 32);
      lrun[r] = lrun[r] * fac[r] + rs;
    }
    // rescale accumulated O
    #pragma unroll
    for (int t = 0; t < 4; ++t)
      #pragma unroll
      for (int r = 0; r < 8; ++r) oacc[t][r] *= fac[r];
    // C/D-layout P -> LDS [16 rows][32 keys] (f16)
    #pragma unroll
    for (int r = 0; r < 8; ++r) {
      psh[wave][(half * 8 + r) * 32 + l16]      = (_Float16)s0[r];
      psh[wave][(half * 8 + r) * 32 + 16 + l16] = (_Float16)s1[r];
    }
    asm volatile("s_wait_dscnt 0" ::: "memory");
    // reload P in A-fragment layout
    v16h pa;
    #pragma unroll
    for (int e = 0; e < 16; ++e)
      pa[e] = psh[wave][l16 * 32 + ((e >> 3) * 16 + half * 8 + (e & 7))];
    // O += P @ V over 4 dim-chunks of 16
    #pragma unroll
    for (int t = 0; t < 4; ++t) {
      v16h vf;
      #pragma unroll
      for (int e = 0; e < 16; ++e)
        vf[e] = Vc[(rowBase + kBase + (e >> 3) * 16 + half * 8 + (e & 7)) * H_ + h * HD_ + t * 16 + l16];
      oacc[t] = WMMA_F16(pa, vf, oacc[t]);
    }
  }
  // normalize and write out (f16)
  #pragma unroll
  for (int t = 0; t < 4; ++t)
    #pragma unroll
    for (int r = 0; r < 8; ++r) {
      int m = qt * 16 + half * 8 + r;
      O[(rowBase + m) * H_ + h * HD_ + t * 16 + l16] = (_Float16)(oacc[t][r] / lrun[r]);
    }
}

// ---------------------------------------------------------------------------
// Host launcher
// ---------------------------------------------------------------------------
extern "C" void kernel_launch(void* const* d_in, const int* in_sizes, int n_in,
                              void* d_out, int out_size, void* d_ws, size_t ws_size,
                              hipStream_t stream) {
  const float* embed   = (const float*)d_in[0];
  const float* pose    = (const float*)d_in[1];
  const float* routerw = (const float*)d_in[2];
  const float* wq      = (const float*)d_in[3];
  const float* wk      = (const float*)d_in[4];
  const float* wv      = (const float*)d_in[5];
  const float* wo      = (const float*)d_in[6];
  const float* alns    = (const float*)d_in[7];
  const float* alnb    = (const float*)d_in[8];
  const float* flns    = (const float*)d_in[9];
  const float* flnb    = (const float*)d_in[10];
  const float* wgate   = (const float*)d_in[11];
  const float* wup     = (const float*)d_in[12];
  const float* wdown   = (const float*)d_in[13];
  const float* finls   = (const float*)d_in[14];
  const float* finlb   = (const float*)d_in[15];
  const float* lmw     = (const float*)d_in[16];
  const int*   ids     = (const int*)d_in[17];
  float* out = (float*)d_out;

  char* wsb = (char*)d_ws;
  float* x   = (float*)wsb;      wsb += (size_t)M_ * H_ * 4;
  float* rl  = (float*)wsb;      wsb += (size_t)M_ * 4;
  float* mk  = (float*)wsb;      wsb += (size_t)M_ * 4;
  _Float16* xn16 = (_Float16*)wsb; wsb += (size_t)M_ * H_ * 2;
  _Float16* q16  = (_Float16*)wsb; wsb += (size_t)M_ * H_ * 2;
  _Float16* k16  = (_Float16*)wsb; wsb += (size_t)M_ * H_ * 2;
  _Float16* v16  = (_Float16*)wsb; wsb += (size_t)M_ * H_ * 2;
  _Float16* ao16 = (_Float16*)wsb; wsb += (size_t)M_ * H_ * 2;
  _Float16* g16  = (_Float16*)wsb; wsb += (size_t)M_ * I_ * 2;
  _Float16* u16  = (_Float16*)wsb; wsb += (size_t)M_ * I_ * 2;
  (void)ws_size; (void)n_in; (void)in_sizes; (void)out_size;

  {
    size_t tot = (size_t)M_ * H_;
    embed_k<<<(int)((tot + 255) / 256), 256, 0, stream>>>(embed, pose, ids, x);
  }

  for (int l = 0; l < L_; ++l) {
    rowdot_k<<<M_, 256, 0, stream>>>(x, routerw + (size_t)l * H_, rl);
    topk_mask_k<<<(M_ + 255) / 256, 256, 0, stream>>>(rl, mk);
    layernorm_k<<<M_, 256, 0, stream>>>(x, alns + (size_t)l * H_, alnb + (size_t)l * H_, xn16);
    GEMM_LAUNCH(H_, H_, false, true, xn16, wq + (size_t)l * H_ * H_, nullptr, q16, nullptr);
    GEMM_LAUNCH(H_, H_, false, true, xn16, wk + (size_t)l * H_ * H_, nullptr, k16, nullptr);
    GEMM_LAUNCH(H_, H_, false, true, xn16, wv + (size_t)l * H_ * H_, nullptr, v16, nullptr);
    flash_attn_k<<<(B_ * NH_ * (S_ / 16)) / 4, 128, 0, stream>>>(q16, k16, v16, ao16);
    GEMM_LAUNCH(H_, H_, true, false, ao16, wo + (size_t)l * H_ * H_, x, nullptr, mk);  // x += mask*o
    layernorm_k<<<M_, 256, 0, stream>>>(x, flns + (size_t)l * H_, flnb + (size_t)l * H_, xn16);
    GEMM_LAUNCH(I_, H_, false, true, xn16, wgate + (size_t)l * H_ * I_, nullptr, g16, nullptr);
    GEMM_LAUNCH(I_, H_, false, true, xn16, wup + (size_t)l * H_ * I_, nullptr, u16, nullptr);
    {
      int tot = M_ * I_;
      swiglu_k<<<(tot + 255) / 256, 256, 0, stream>>>(g16, u16, g16, tot);
    }
    GEMM_LAUNCH(H_, I_, true, false, g16, wdown + (size_t)l * I_ * H_, x, nullptr, mk); // x += mask*f
  }

  layernorm_k<<<M_, 256, 0, stream>>>(x, finls, finlb, xn16);
  GEMM_LAUNCH(V_, H_, false, false, xn16, lmw, out, nullptr, nullptr);   // logits [B*S, V]
}
